// Qwen3Attention_38173669327170
// MI455X (gfx1250) — matrix operand order repaired and checked by reading, not compiled, not run on hardware
//
#include <hip/hip_runtime.h>

// ---------- types ----------
typedef __attribute__((ext_vector_type(16))) __bf16        v16bf;
typedef __attribute__((ext_vector_type(8)))  float         v8f;
typedef __attribute__((ext_vector_type(4)))  unsigned int  u32x4;
typedef int v4i_vec __attribute__((vector_size(16)));   // matches builtin param type

#ifndef __has_builtin
#define __has_builtin(x) 0
#endif
#if __has_builtin(__builtin_amdgcn_global_load_async_to_lds_b128)
#define HAVE_ASYNC 1
#else
#define HAVE_ASYNC 0
#endif

// RNE float -> bf16
static __device__ inline unsigned short f2bf(float f) {
  unsigned int u = __float_as_uint(f);
  u += 0x7FFFu + ((u >> 16) & 1u);
  return (unsigned short)(u >> 16);
}
static __device__ inline unsigned int pk2(float lo, float hi) {
  return (unsigned int)f2bf(lo) | ((unsigned int)f2bf(hi) << 16);
}
// Build a 16-element bf16 fragment from two 16-byte chunks
static __device__ inline v16bf ldfrag(const unsigned short* p0, const unsigned short* p1) {
  union { u32x4 q[2]; v16bf v; } u;
  u.q[0] = *(const u32x4*)p0;
  u.q[1] = *(const u32x4*)p1;
  return u.v;
}
static __device__ inline v8f wmma_bf16(v16bf a, v16bf b, v8f c) {
  return __builtin_amdgcn_wmma_f32_16x16x32_bf16(false, a, false, b, (short)0, c, false, false);
}
// 16B global -> LDS copy; async path on CDNA5 (ASYNCcnt tracked)
static __device__ inline void async_cp16(const unsigned short* g, unsigned short* l) {
#if HAVE_ASYNC
  __builtin_amdgcn_global_load_async_to_lds_b128(
      (__attribute__((address_space(1))) v4i_vec*)g,
      (__attribute__((address_space(3))) v4i_vec*)l, 0, 0);
#else
  *(u32x4*)l = *(const u32x4*)g;
#endif
}

#define B_   2
#define S_   1024
#define HID_ 4096
#define NH_  32
#define NKV_ 8
#define HD_  128
#define SCALE_ 0.08838834764831845f

// ============================================================================
// Kernel 0: elementwise fp32 -> bf16 (weights/activations), 8 elems/thread
// ============================================================================
__global__ __launch_bounds__(256)
void convert_f32_bf16(const float* __restrict__ src, unsigned short* __restrict__ dst) {
  const size_t i = (size_t)blockIdx.x * 256 + threadIdx.x;
  const float4* s = (const float4*)src + i * 2;
  float4 a = s[0], b = s[1];
  ((u32x4*)dst)[i] = (u32x4){pk2(a.x, a.y), pk2(a.z, a.w), pk2(b.x, b.y), pk2(b.z, b.w)};
}

// ============================================================================
// Kernel 1/4: GEMM  C[M,N] = A[M,K] * W[N,K]^T   (bf16 in, fp32 out)
// 128x64 block tile / 4 waves (each 32x64 = 8 WMMA per k-step);
// double-buffered LDS fed by GLOBAL_LOAD_ASYNC_TO_LDS_B128 + s_wait_asynccnt
// ============================================================================
static __device__ inline void stage_tiles(const unsigned short* __restrict__ Ab,
                                          const unsigned short* __restrict__ Wb,
                                          unsigned short* lA, unsigned short* lB,
                                          int t, int K) {
  // A tile 128x32 bf16 = 512 16B chunks; B tile 64x32 = 256 chunks; 128 threads
#pragma unroll
  for (int i = 0; i < 4; ++i) {
    const int c = t + 128 * i, row = c >> 2, col = (c & 3) * 8;
    async_cp16(Ab + (size_t)row * K + col, lA + row * 32 + col);
  }
#pragma unroll
  for (int i = 0; i < 2; ++i) {
    const int c = t + 128 * i, row = c >> 2, col = (c & 3) * 8;
    async_cp16(Wb + (size_t)row * K + col, lB + row * 32 + col);
  }
}

__global__ __launch_bounds__(128)
void gemm_bf16_wmma(const unsigned short* __restrict__ A,
                    const unsigned short* __restrict__ W,
                    float* __restrict__ C, int M, int N, int K) {
  __shared__ __align__(16) unsigned short lsA[2][128 * 32];
  __shared__ __align__(16) unsigned short lsB[2][64 * 32];

  const int t  = threadIdx.x;
  const int m0 = blockIdx.y * 128;
  const int n0 = blockIdx.x * 64;
  const int w    = t >> 5;
  const int lane = t & 31;
  const int lm   = lane & 15;
  const int hi   = (lane >> 4) & 1;
  const int ka   = hi ? 8 : 0;    // A-frag K chunk base
  const int kb   = hi ? 8 : 0;    // B-frag K chunk base

  const unsigned short* Ab = A + (size_t)m0 * K;
  const unsigned short* Wb = W + (size_t)n0 * K;

  v8f acc[2][4];
#pragma unroll
  for (int s2 = 0; s2 < 2; ++s2)
#pragma unroll
    for (int j = 0; j < 4; ++j) acc[s2][j] = (v8f){0,0,0,0,0,0,0,0};

  stage_tiles(Ab, Wb, lsA[0], lsB[0], t, K);
  int bb = 0;
  for (int k0 = 0; k0 < K; k0 += 32) {
    if (k0 + 32 < K) {
      stage_tiles(Ab + k0 + 32, Wb + k0 + 32, lsA[bb ^ 1], lsB[bb ^ 1], t, K);
#if HAVE_ASYNC
      asm volatile("s_wait_asynccnt 0x6" ::: "memory");  // previous stage done
#endif
    } else {
#if HAVE_ASYNC
      asm volatile("s_wait_asynccnt 0x0" ::: "memory");
#endif
    }
    __syncthreads();
    const unsigned short* la = lsA[bb];
    const unsigned short* lb = lsB[bb];
    // A fragments: lane = M row; low half K 0..7 & 16..23, high K 8..15 & 24..31
    const unsigned short* pa0 = &la[(w * 32 + lm) * 32 + ka];
    const unsigned short* pa1 = &la[(w * 32 + 16 + lm) * 32 + ka];
    v16bf af0 = ldfrag(pa0, pa0 + 16);
    v16bf af1 = ldfrag(pa1, pa1 + 16);
#pragma unroll
    for (int j = 0; j < 4; ++j) {
      // B fragment: lane = N col; low half K 0..7 & 16..23, high K 8..15 & 24..31
      const unsigned short* pb = &lb[(j * 16 + lm) * 32 + kb];
      v16bf bf = ldfrag(pb, pb + 16);
      acc[0][j] = wmma_bf16(af0, bf, acc[0][j]);
      acc[1][j] = wmma_bf16(af1, bf, acc[1][j]);
    }
    __syncthreads();   // all waves done with buffer bb before it is restaged
    bb ^= 1;
  }
#pragma unroll
  for (int s2 = 0; s2 < 2; ++s2)
#pragma unroll
    for (int j = 0; j < 4; ++j) {
      float* cp = C + (size_t)(m0 + w * 32 + s2 * 16 + hi * 8) * N + n0 + j * 16 + lm;
#pragma unroll
      for (int e = 0; e < 8; ++e) cp[(size_t)e * N] = acc[s2][j][e];
    }
}

// ============================================================================
// Kernel 2/4: per-head RMSNorm + RoPE; emits bf16 qhat/khat and transposed vT
// ============================================================================
__global__ __launch_bounds__(256)
void norm_rope(const float* __restrict__ qkv, const float* __restrict__ cosb,
               const float* __restrict__ sinb, const float* __restrict__ wq,
               const float* __restrict__ wk, unsigned short* __restrict__ qhat,
               unsigned short* __restrict__ khat, unsigned short* __restrict__ vT) {
  const int wid  = blockIdx.x * 8 + (threadIdx.x >> 5);
  const int lane = threadIdx.x & 31;
  const int m = wid / 48;          // token index b*S + s
  const int c = wid - m * 48;      // 0..31 Q heads, 32..39 K heads, 40..47 V heads
  const int b = m >> 10, s = m & 1023;

  if (c < 40) {
    const int h = (c < 32) ? c : c - 32;
    const float* x = qkv + (size_t)m * 6144 + ((c < 32) ? h * 128 : 4096 + h * 128);
    float x0 = x[lane], x1 = x[lane + 32], x2 = x[lane + 64], x3 = x[lane + 96];
    float ss = x0*x0 + x1*x1 + x2*x2 + x3*x3;
#pragma unroll
    for (int mk = 1; mk < 32; mk <<= 1) ss += __shfl_xor(ss, mk);
    const float rn = rsqrtf(ss * (1.0f / 128.0f) + 1e-6f);
    const float* wn = (c < 32) ? wq : wk;
    x0 *= rn * wn[lane];      x1 *= rn * wn[lane + 32];
    x2 *= rn * wn[lane + 64]; x3 *= rn * wn[lane + 96];
    const float* cp = cosb + (size_t)m * 128;
    const float* sp = sinb + (size_t)m * 128;
    float o0 = x0 * cp[lane]      - x2 * sp[lane];
    float o1 = x1 * cp[lane + 32] - x3 * sp[lane + 32];
    float o2 = x2 * cp[lane + 64] + x0 * sp[lane + 64];
    float o3 = x3 * cp[lane + 96] + x1 * sp[lane + 96];
    const float sc = (c < 32) ? SCALE_ : 1.0f;   // fold attention scale into q
    unsigned short* dst = (c < 32)
        ? qhat + ((size_t)(b * NH_  + h) * S_ + s) * HD_
        : khat + ((size_t)(b * NKV_ + h) * S_ + s) * HD_;
    dst[lane]      = f2bf(o0 * sc); dst[lane + 32] = f2bf(o1 * sc);
    dst[lane + 64] = f2bf(o2 * sc); dst[lane + 96] = f2bf(o3 * sc);
  } else {
    const int kvh = c - 40;
    const float* x = qkv + (size_t)m * 6144 + 5120 + kvh * 128;
    unsigned short* dst = vT + ((size_t)(b * NKV_ + kvh) * HD_) * S_ + s;
#pragma unroll
    for (int i = 0; i < 4; ++i)
      dst[(size_t)(lane + i * 32) * S_] = f2bf(x[lane + i * 32]);
  }
}

// ============================================================================
// Kernel 3/4: flash attention (non-causal, mask is all-ones), bf16 output
// ============================================================================
__global__ __launch_bounds__(128)
void attn_fa(const unsigned short* __restrict__ qhat,
             const unsigned short* __restrict__ khat,
             const unsigned short* __restrict__ vT,
             unsigned short* __restrict__ out) {
  __shared__ __align__(16) unsigned short Pl[4][16 * 32];

  const int b = blockIdx.z, h = blockIdx.y;
  const int w = threadIdx.x >> 5, lane = threadIdx.x & 31;
  const int lm = lane & 15, hi = (lane >> 4) & 1;
  const int q0 = blockIdx.x * 64 + w * 16;
  const int kv = h >> 2;   // NREP = 4

  const unsigned short* qp = qhat + ((size_t)(b * NH_  + h)  * S_ + q0) * HD_;
  const unsigned short* kp = khat + ((size_t)(b * NKV_ + kv) * S_) * HD_;
  const unsigned short* vp = vT   + ((size_t)(b * NKV_ + kv) * HD_) * S_;

  v16bf qf[4];
#pragma unroll
  for (int f = 0; f < 4; ++f) {
    const unsigned short* p = qp + (size_t)lm * HD_ + f * 32 + hi * 8;
    qf[f] = ldfrag(p, p + 16);
  }

  v8f O[8];
  float rm[8], rs[8];
#pragma unroll
  for (int t2 = 0; t2 < 8; ++t2) O[t2] = (v8f){0,0,0,0,0,0,0,0};
#pragma unroll
  for (int j = 0; j < 8; ++j) { rm[j] = -1e30f; rs[j] = 0.0f; }

  for (int k0 = 0; k0 < S_; k0 += 32) {
    v8f S0 = (v8f){0,0,0,0,0,0,0,0};
    v8f S1 = (v8f){0,0,0,0,0,0,0,0};
#pragma unroll
    for (int f = 0; f < 4; ++f) {
      const unsigned short* pk0 = kp + (size_t)(k0 + lm) * HD_ + f * 32 + hi * 8;
      S0 = wmma_bf16(qf[f], ldfrag(pk0, pk0 + 16), S0);
      const unsigned short* pk1 = kp + (size_t)(k0 + 16 + lm) * HD_ + f * 32 + hi * 8;
      S1 = wmma_bf16(qf[f], ldfrag(pk1, pk1 + 16), S1);
    }
    float al[8];
#pragma unroll
    for (int j = 0; j < 8; ++j) {
      float mj = fmaxf(S0[j], S1[j]);
      mj = fmaxf(mj, __shfl_xor(mj, 1));
      mj = fmaxf(mj, __shfl_xor(mj, 2));
      mj = fmaxf(mj, __shfl_xor(mj, 4));
      mj = fmaxf(mj, __shfl_xor(mj, 8));
      const float nm = fmaxf(rm[j], mj);
      const float a  = __expf(rm[j] - nm);
      rm[j] = nm;
      const float p0 = __expf(S0[j] - nm);
      const float p1 = __expf(S1[j] - nm);
      S0[j] = p0; S1[j] = p1;
      float sj = p0 + p1;
      sj += __shfl_xor(sj, 1);
      sj += __shfl_xor(sj, 2);
      sj += __shfl_xor(sj, 4);
      sj += __shfl_xor(sj, 8);
      rs[j] = rs[j] * a + sj;
      al[j] = a;
    }
#pragma unroll
    for (int t2 = 0; t2 < 8; ++t2)
#pragma unroll
      for (int j = 0; j < 8; ++j) O[t2][j] *= al[j];

    // transpose P through wave-private LDS into A-fragment layout
    unsigned short* pw = Pl[w];
#pragma unroll
    for (int j = 0; j < 8; ++j) {
      const int r = j + hi * 8;
      pw[r * 32 + lm]      = f2bf(S0[j]);
      pw[r * 32 + 16 + lm] = f2bf(S1[j]);
    }
    const unsigned short* pr = pw + lm * 32 + (hi ? 8 : 0);
    v16bf pf = ldfrag(pr, pr + 16);
#pragma unroll
    for (int t2 = 0; t2 < 8; ++t2) {
      const unsigned short* vb = vp + (size_t)(t2 * 16 + lm) * S_ + k0 + hi * 8;
      O[t2] = wmma_bf16(pf, ldfrag(vb, vb + 16), O[t2]);
    }
  }
#pragma unroll
  for (int j = 0; j < 8; ++j) {
    const float inv = 1.0f / rs[j];
    const int row = q0 + j + hi * 8;
    unsigned short* op = out + ((size_t)(b * S_) + row) * (NH_ * HD_) + h * HD_;
#pragma unroll
    for (int t2 = 0; t2 < 8; ++t2) op[t2 * 16 + lm] = f2bf(O[t2][j] * inv);
  }
}

// ============================================================================
// launch
// ============================================================================
extern "C" void kernel_launch(void* const* d_in, const int* in_sizes, int n_in,
                              void* d_out, int out_size, void* d_ws, size_t ws_size,
                              hipStream_t stream) {
  const float* hidden = (const float*)d_in[0];
  const float* cosb   = (const float*)d_in[1];
  const float* sinb   = (const float*)d_in[2];
  // d_in[3] = attention_mask (all ones) -> ignored
  const float* w_qkv  = (const float*)d_in[4];
  const float* w_qn   = (const float*)d_in[5];
  const float* w_kn   = (const float*)d_in[6];
  const float* w_o    = (const float*)d_in[7];

  char* ws = (char*)d_ws;
  unsigned short* hbf    = (unsigned short*)(ws);                    // 16777216
  unsigned short* wqkvbf = (unsigned short*)(ws + 16777216);         // 50331648
  unsigned short* wobf   = (unsigned short*)(ws + 67108864);         // 33554432
  float*          qkv    = (float*)(ws + 100663296);                 // 50331648
  unsigned short* qhat   = (unsigned short*)(ws + 150994944);        // 16777216
  unsigned short* khat   = (unsigned short*)(ws + 167772160);        // 4194304
  unsigned short* vT     = (unsigned short*)(ws + 171966464);        // 4194304
  unsigned short* aoutbf = (unsigned short*)(ws + 176160768);        // 16777216

  const int M = B_ * S_;  // 2048
  // 0) one-shot fp32 -> bf16 conversions (weights fit in 192MB L2 afterwards)
  convert_f32_bf16<<<(M * HID_) / (8 * 256), 256, 0, stream>>>(hidden, hbf);
  convert_f32_bf16<<<((NH_ + 2 * NKV_) * HD_ * HID_) / (8 * 256), 256, 0, stream>>>(w_qkv, wqkvbf);
  convert_f32_bf16<<<(HID_ * HID_) / (8 * 256), 256, 0, stream>>>(w_o, wobf);
  // 1) QKV projection: [2048,4096] x [6144,4096]^T
  gemm_bf16_wmma<<<dim3((NH_ + 2 * NKV_) * HD_ / 64, M / 128), 128, 0, stream>>>(
      hbf, wqkvbf, qkv, M, (NH_ + 2 * NKV_) * HD_, HID_);
  // 2) RMSNorm + RoPE + layout conversion
  norm_rope<<<(M * (NH_ + 2 * NKV_)) / 8, 256, 0, stream>>>(
      qkv, cosb, sinb, w_qn, w_kn, qhat, khat, vT);
  // 3) flash attention (writes bf16 for the final GEMM)
  attn_fa<<<dim3(S_ / 64, NH_, B_), 128, 0, stream>>>(qhat, khat, vT, aoutbf);
  // 4) output projection: [2048,4096] x [4096,4096]^T
  gemm_bf16_wmma<<<dim3(HID_ / 64, M / 128), 128, 0, stream>>>(
      aoutbf, wobf, (float*)d_out, M, HID_, HID_);
}
